// GraphAttentionConv_51513837748779
// MI455X (gfx1250) — matrix operand order repaired and checked by reading, not compiled, not run on hardware
//
#include <hip/hip_runtime.h>
#include <hip/hip_bf16.h>

typedef __attribute__((ext_vector_type(16))) __bf16 v16bf;
typedef __attribute__((ext_vector_type(8)))  __bf16 v8bf;
typedef __attribute__((ext_vector_type(8)))  float  v8f;
typedef __attribute__((ext_vector_type(4)))  float  v4f;
typedef __attribute__((ext_vector_type(4)))  int    v4i;

#define N_NODES 8192
#define FEAT    512
#define ALPHA   0.2f
#define TJ      128   // columns per flash tile

// ---------------------------------------------------------------- fp32 -> bf16
__global__ void cvt_bf16_kernel(const float* __restrict__ in,
                                __bf16* __restrict__ out, int n) {
  int i = blockIdx.x * blockDim.x + threadIdx.x;
  if (i < n) out[i] = (__bf16)in[i];
}

// ---------------------------------------------------------------- GEMM1
// hT[c * N_NODES + r] = sum_k x[r,k] * W[c,k]   (h stored transposed, bf16)
// One workgroup = 16 rows x 512 cols. 8 waves; wave w owns cols [64w, 64w+64).
__global__ __launch_bounds__(256) void gemm_h_kernel(
    const __bf16* __restrict__ xb, const __bf16* __restrict__ wb,
    __bf16* __restrict__ hT) {
  const int t = threadIdx.x;
  const int lane = t & 31, wave = t >> 5;
  const int laneHi = lane >> 4, laneLo = lane & 15;
  const int i0 = blockIdx.x * 16;
  const int colBase = wave * 64;

  v8f acc[4] = {};
  const __bf16* xrow = xb + (size_t)(i0 + laneLo) * FEAT;

  for (int k0 = 0; k0 < FEAT; k0 += 32) {
    // A (16x32 bf16): lanes 0-15 hold K {0..15}, lanes 16-31 K {16..31}
    v8bf lo = *(const v8bf*)(xrow + k0 + 16 * laneHi);
    v8bf hi = *(const v8bf*)(xrow + k0 + 8 + 16 * laneHi);
    v16bf a;
#pragma unroll
    for (int i = 0; i < 8; ++i) { a[i] = lo[i]; a[8 + i] = hi[i]; }
    // batch all B tiles first so loads overlap the WMMA stream
    v16bf b[4];
#pragma unroll
    for (int ct = 0; ct < 4; ++ct) {
      int n = colBase + ct * 16 + laneLo;
      b[ct] = *(const v16bf*)(wb + (size_t)n * FEAT + k0 + 16 * laneHi);
    }
#pragma unroll
    for (int ct = 0; ct < 4; ++ct)
      acc[ct] = __builtin_amdgcn_wmma_f32_16x16x32_bf16(
          false, a, false, b[ct], (short)0, acc[ct], false, false);
  }
#pragma unroll
  for (int ct = 0; ct < 4; ++ct) {
    int c = colBase + ct * 16 + laneLo;
#pragma unroll
    for (int v = 0; v < 8; ++v) {           // C layout: row = v + 8*laneHi
      int r = i0 + v + 8 * laneHi;
      hT[(size_t)c * N_NODES + r] = (__bf16)acc[ct][v];
    }
  }
}

// ---------------------------------------------------------------- src/dst
__global__ void src_dst_kernel(const __bf16* __restrict__ hT,
                               const float* __restrict__ a1,
                               const float* __restrict__ a2,
                               float* __restrict__ srcv,
                               float* __restrict__ dstv) {
  int i = blockIdx.x * blockDim.x + threadIdx.x;   // node index (coalesced)
  float s = 0.f, d = 0.f;
  for (int k = 0; k < FEAT; ++k) {
    float hv = (float)hT[(size_t)k * N_NODES + i];
    s = fmaf(hv, a1[k], s);
    d = fmaf(hv, a2[k], d);
  }
  srcv[i] = s;
  dstv[i] = d;
}

// ---------------------------------------------------------------- fused flash GAT
// WG = 16 rows; tiles of 16x128 over all 8192 columns. Scores live in registers,
// exp is fully parallel across 256 threads, row max/sum via 16x16 LDS partials.
// acc (per wave: 16 rows x 64 feats, 4 WMMA C tiles) rescaled once per 128 cols.
__global__ __launch_bounds__(256) void gat_flash_kernel(
    const int* __restrict__ adj, const __bf16* __restrict__ hT,
    const float* __restrict__ srcv, const float* __restrict__ dstv,
    float* __restrict__ out) {
  __shared__ __attribute__((aligned(32))) __bf16 p_tile[16 * TJ];  // 4 KB
  __shared__ float red[16][16];
  __shared__ float sh_m[16], sh_l[16], sh_scale[16], sh_newm[16];

  const int t = threadIdx.x;
  const int lane = t & 31, wave = t >> 5;
  const int laneHi = lane >> 4, laneLo = lane & 15;
  const int r = t >> 4;          // row 0..15 (softmax mapping)
  const int c = t & 15;          // column group (8 contiguous cols each)
  const int i0 = blockIdx.x * 16;
  const float NEG = -__builtin_inff();

  if (t < 16) { sh_m[t] = NEG; sh_l[t] = 0.f; }
  const float srow = srcv[i0 + r];
  v8f acc[4] = {};
  __syncthreads();

  for (int j0 = 0; j0 < N_NODES; j0 += TJ) {
    // ---- scores for this thread's 8 contiguous columns (vector adj/dst loads)
    const int jb = j0 + c * 8;
    const int* ap = adj + (size_t)(i0 + r) * N_NODES + jb;
    v4i am0 = *(const v4i*)(ap);
    v4i am1 = *(const v4i*)(ap + 4);
    if (j0 + TJ < N_NODES) __builtin_prefetch(ap + TJ, 0, 0);
    v4f d0 = *(const v4f*)(dstv + jb);
    v4f d1 = *(const v4f*)(dstv + jb + 4);
    float s[8];
    float lmax = NEG;
#pragma unroll
    for (int q = 0; q < 8; ++q) {
      float sv = srow + (q < 4 ? d0[q] : d1[q - 4]);
      sv = (sv > 0.f) ? sv : ALPHA * sv;
      int m = (q < 4 ? am0[q] : am1[q - 4]);
      sv = m ? sv : NEG;
      s[q] = sv;
      lmax = fmaxf(lmax, sv);
    }
    red[r][c] = lmax;
    __syncthreads();

    // ---- row max + rescale factor (16 threads, 16-iter reduce)
    if (c == 0) {
      float tmax = red[r][0];
#pragma unroll
      for (int q = 1; q < 16; ++q) tmax = fmaxf(tmax, red[r][q]);
      float oldm = sh_m[r];
      float newm = fmaxf(oldm, tmax);
      sh_newm[r]  = newm;
      sh_scale[r] = (newm == NEG) ? 1.f : __expf(oldm - newm);  // exp(-inf)=0 ok
      sh_m[r]     = newm;
    }
    __syncthreads();

    // ---- p = exp(s - newm): fully parallel across 256 threads
    float newm = sh_newm[r];
    float psum = 0.f;
#pragma unroll
    for (int q = 0; q < 8; ++q) {
      float p = (s[q] == NEG) ? 0.f : __expf(s[q] - newm);
      p_tile[r * TJ + c * 8 + q] = (__bf16)p;
      psum += p;
    }
    red[r][c] = psum;
    __syncthreads();

    // ---- l update (overlaps with WMMA below; touches only sh_l/red)
    if (c == 0) {
      float ps = 0.f;
#pragma unroll
      for (int q = 0; q < 16; ++q) ps += red[r][q];
      sh_l[r] = sh_l[r] * sh_scale[r] + ps;
    }

    // ---- rescale accumulators once per 128 cols, then 16 WMMAs
    float rs[8];
#pragma unroll
    for (int v = 0; v < 8; ++v) rs[v] = sh_scale[v + 8 * laneHi];
#pragma unroll
    for (int ct = 0; ct < 4; ++ct)
#pragma unroll
      for (int v = 0; v < 8; ++v) acc[ct][v] *= rs[v];

#pragma unroll
    for (int kk = 0; kk < TJ; kk += 32) {
      const __bf16* prow = p_tile + laneLo * TJ + kk;   // A from LDS p tile
      v8bf plo = *(const v8bf*)(prow + 16 * laneHi);
      v8bf phi = *(const v8bf*)(prow + 8 + 16 * laneHi);
      v16bf a;
#pragma unroll
      for (int i = 0; i < 8; ++i) { a[i] = plo[i]; a[8 + i] = phi[i]; }
      v16bf b[4];
#pragma unroll
      for (int ct = 0; ct < 4; ++ct) {
        int col = wave * 64 + ct * 16 + laneLo;
        // contiguous 16 K-values of column `col` (hT is L2-resident, 8 MB)
        b[ct] = *(const v16bf*)(hT + (size_t)col * N_NODES + j0 + kk + 16 * laneHi);
      }
#pragma unroll
      for (int ct = 0; ct < 4; ++ct)
        acc[ct] = __builtin_amdgcn_wmma_f32_16x16x32_bf16(
            false, a, false, b[ct], (short)0, acc[ct], false, false);
    }
    __syncthreads();
  }

  // ---- normalize by row sum and store
  if (t < 16) sh_scale[t] = (sh_l[t] > 0.f) ? (1.f / sh_l[t]) : 0.f;
  __syncthreads();
  float inv[8];
#pragma unroll
  for (int v = 0; v < 8; ++v) inv[v] = sh_scale[v + 8 * laneHi];
#pragma unroll
  for (int ct = 0; ct < 4; ++ct) {
    int col = wave * 64 + ct * 16 + laneLo;
#pragma unroll
    for (int v = 0; v < 8; ++v) {
      int rr = i0 + v + 8 * laneHi;
      out[(size_t)rr * FEAT + col] = acc[ct][v] * inv[v];
    }
  }
}

// ---------------------------------------------------------------- launch
extern "C" void kernel_launch(void* const* d_in, const int* in_sizes, int n_in,
                              void* d_out, int out_size, void* d_ws, size_t ws_size,
                              hipStream_t stream) {
  const float* x   = (const float*)d_in[0];
  const int*   adj = (const int*)d_in[1];
  const float* W   = (const float*)d_in[2];
  const float* a1  = (const float*)d_in[3];
  const float* a2  = (const float*)d_in[4];
  float* out = (float*)d_out;

  char* ws = (char*)d_ws;
  const size_t MB = 1024 * 1024;
  __bf16* x_bf  = (__bf16*)(ws);                 // 8192*512*2  = 8 MiB
  __bf16* w_bf  = (__bf16*)(ws + 8 * MB);        // 512*512*2   = 0.5 MiB
  __bf16* h_bfT = (__bf16*)(ws + 9 * MB);        // 512*8192*2  = 8 MiB (transposed h)
  float*  srcv  = (float*)(ws + 17 * MB);        // 8192*4
  float*  dstv  = (float*)(ws + 17 * MB + 64 * 1024);

  const int nx = N_NODES * FEAT;
  const int nw = FEAT * FEAT;
  cvt_bf16_kernel<<<(nx + 255) / 256, 256, 0, stream>>>(x, x_bf, nx);
  cvt_bf16_kernel<<<(nw + 255) / 256, 256, 0, stream>>>(W, w_bf, nw);
  gemm_h_kernel<<<N_NODES / 16, 256, 0, stream>>>(x_bf, w_bf, h_bfT);
  src_dst_kernel<<<N_NODES / 256, 256, 0, stream>>>(h_bfT, a1, a2, srcv, dstv);
  gat_flash_kernel<<<N_NODES / 16, 256, 0, stream>>>(adj, h_bfT, srcv, dstv, out);
}
